// ModuleRelProp_24790551232616
// MI455X (gfx1250) — compile-verified, matrix operand order checked by reading, not run on hardware
//
#include <hip/hip_runtime.h>
#include <hip/hip_bf16.h>

// Problem constants (match reference)
#define Bb 4
#define Mm 1024
#define Nn 256
#define Dd 512
#define Hh 256
#define Ll 32

#define GK  512   // K for all dense GEMMs (proj + gate): compile-time
#define GKT 16    // GK / 32

typedef __bf16 bf16;
typedef __attribute__((ext_vector_type(2)))  bf16  v2bf;
typedef __attribute__((ext_vector_type(8)))  bf16  v8bf;
typedef __attribute__((ext_vector_type(16))) bf16  v16bf;
typedef __attribute__((ext_vector_type(8)))  float v8f;

// ---------------------------------------------------------------------------
// WMMA helpers (CDNA5 gfx1250, wave32): D = A(16x32 bf16) * B(32x16 bf16) + C
// ---------------------------------------------------------------------------
__device__ __forceinline__ v8f wmma_bf16(v16bf a, v16bf b, v8f c) {
  return __builtin_amdgcn_wmma_f32_16x16x32_bf16(
      /*neg_a=*/false, a, /*neg_b=*/false, b,
      /*c_mod=*/(short)0, c, /*reuse_a=*/false, /*reuse_b=*/false);
}

__device__ __forceinline__ v8f zero8() {
  v8f z;
#pragma unroll
  for (int i = 0; i < 8; ++i) z[i] = 0.0f;
  return z;
}

// p[0..7] and p[16..23] -> v16bf via two 16B vector loads (works for LDS/global).
__device__ __forceinline__ v16bf load2x8(const bf16* p) {
  v8bf lo = *(const v8bf*)(p);
  v8bf hi = *(const v8bf*)(p + 16);
  return __builtin_shufflevector(lo, hi, 0, 1, 2, 3, 4, 5, 6, 7,
                                 8, 9, 10, 11, 12, 13, 14, 15);
}

// A operand (16 rows x 32 K) from a row-major tile with row stride ld
// (ld must be a multiple of 8 and the base 16B-aligned -> 2x b128 per lane).
// Lane t: row t&15; K = {half*8 + e} and {16 + half*8 + e}.
__device__ __forceinline__ v16bf load_a_rm(const bf16* base, int ld, int lane) {
  return load2x8(base + (lane & 15) * ld + ((lane >> 4) << 3));
}

// ---------------------------------------------------------------------------
// Pack a row-major f32 matrix [K x Nc] (batched via blockIdx.y) into WMMA
// B-operand tile order: tile(nt,kt) at (nt*(K/32)+kt)*512; within a tile,
// value (kk,nn) sits at lane = nn + 16*((kk>>3)&1), elem = (kk&7) + 8*(kk>>4).
// Consecutive kt tiles are contiguous -> K-loop walks immediate offsets.
// ---------------------------------------------------------------------------
__global__ void swizzle_b_kernel(const float* __restrict__ src, bf16* __restrict__ dst,
                                 int K, int Nc) {
  int total = K * Nc;
  int idx = blockIdx.x * 256 + threadIdx.x;
  if (idx >= total) return;
  const float* s = src + (size_t)blockIdx.y * total;
  bf16* d = dst + (size_t)blockIdx.y * total;
  int tile = idx >> 9, r = idx & 511;
  int lane = r >> 4, elem = r & 15;
  int ktcnt = K >> 5;
  int nt = tile / ktcnt, kt = tile - nt * ktcnt;
  int nn = lane & 15;
  int kk = ((lane >> 4) << 3) + (elem & 7) + ((elem >> 3) << 4);
  d[idx] = (bf16)s[(kt * 32 + kk) * Nc + nt * 16 + nn];
}

// ---------------------------------------------------------------------------
// Small elementwise kernels
// ---------------------------------------------------------------------------
__global__ void pack_update_kernel(const float* __restrict__ src, float* __restrict__ dstf,
                                   bf16* __restrict__ dstb, int n) {
  int i = blockIdx.x * 256 + threadIdx.x;
  if (i < n) {
    float v = src[i];
    dstf[i] = v;
    dstb[i] = (bf16)v;
  }
}

__global__ void copy_f32_kernel(const float* __restrict__ src, float* __restrict__ dst, int n) {
  int i = blockIdx.x * 256 + threadIdx.x;
  if (i < n) dst[i] = src[i];
}

__global__ void gate_finish_kernel(const float* __restrict__ gacc, const float* __restrict__ b_gate,
                                   const float* __restrict__ u, const float* __restrict__ cx,
                                   float* __restrict__ updOut, bf16* __restrict__ updbOut, int total) {
  int idx = blockIdx.x * 256 + threadIdx.x;
  if (idx >= total) return;
  int d = idx & (Dd - 1);
  float x  = gacc[idx] + b_gate[d];
  float g  = 1.0f / (1.0f + __expf(-x));
  float nu = g * u[idx] + (1.0f - g) * cx[idx];
  updOut[idx]  = nu;
  updbOut[idx] = (bf16)nu;
}

// overwrite_spans scatter: out_all[b, prune[b,n], :] = upd[b,n,:] for n < len[b]
__global__ void scatter_kernel(float* __restrict__ out_all, const float* __restrict__ upd,
                               const int* __restrict__ prune, const int* __restrict__ lens) {
  int bn = blockIdx.x;
  int b = bn >> 8, n = bn & 255;
  if (n >= lens[b]) return;
  int m = prune[bn];
  const float* src = upd + bn * Dd;
  float* dst = out_all + (b * Mm + m) * Dd;
  for (int d = threadIdx.x; d < Dd; d += 256) dst[d] = src[d];
}

// ---------------------------------------------------------------------------
// LDS-free bf16 WMMA GEMM specialized for K=512:
//   C[M,N] (+)= A[M,512] @ Bsw + bias.
// A row-major bf16; Bsw pre-swizzled tiles (contiguous over kt -> immediate
// offsets). Software-pipelined: operands for kt+1 are in flight while kt's
// WMMAs issue, so the compiler can emit partial loadcnt waits.
// Block 256 thr, tile 128(M) x 64(N); requires M%128==0, N%64==0.
// ---------------------------------------------------------------------------
__global__ __launch_bounds__(256) void gemm_bf16_k512_kernel(
    const bf16* __restrict__ A, const bf16* __restrict__ Bsw,
    float* __restrict__ C, const float* __restrict__ bias,
    int M, int N, int accumulate) {
  int t = threadIdx.x, w = t >> 5, lane = t & 31;
  int m0 = blockIdx.y * 128, n0 = blockIdx.x * 64;

  v8f acc[4];
#pragma unroll
  for (int ct = 0; ct < 4; ++ct) acc[ct] = zero8();

  const bf16* arow = A + (size_t)(m0 + w * 16 + (lane & 15)) * GK + ((lane >> 4) << 3);
  // base of this block's B tiles; ct stride = GKT tiles, kt stride = 1 tile
  const bf16* bbase = Bsw + ((size_t)(n0 >> 4) * GKT << 9) + lane * 16;
#define BT(ct, kt) (bbase + (ct) * (GKT << 9) + (kt) * 512)

  v16bf avc = load2x8(arow);
  v16bf bc0 = *(const v16bf*)BT(0, 0);
  v16bf bc1 = *(const v16bf*)BT(1, 0);
  v16bf bc2 = *(const v16bf*)BT(2, 0);
  v16bf bc3 = *(const v16bf*)BT(3, 0);

#pragma unroll
  for (int kt = 0; kt < GKT; ++kt) {
    v16bf avn = avc, bn0 = bc0, bn1 = bc1, bn2 = bc2, bn3 = bc3;
    if (kt + 1 < GKT) {  // resolved at compile time (full unroll)
      avn = load2x8(arow + (kt + 1) * 32);
      bn0 = *(const v16bf*)BT(0, kt + 1);
      bn1 = *(const v16bf*)BT(1, kt + 1);
      bn2 = *(const v16bf*)BT(2, kt + 1);
      bn3 = *(const v16bf*)BT(3, kt + 1);
    }
    acc[0] = wmma_bf16(avc, bc0, acc[0]);
    acc[1] = wmma_bf16(avc, bc1, acc[1]);
    acc[2] = wmma_bf16(avc, bc2, acc[2]);
    acc[3] = wmma_bf16(avc, bc3, acc[3]);
    avc = avn; bc0 = bn0; bc1 = bn1; bc2 = bn2; bc3 = bn3;
  }
#undef BT

  int nl = lane & 15, half = lane >> 4;
#pragma unroll
  for (int ct = 0; ct < 4; ++ct) {
    int n = n0 + ct * 16 + nl;
    float bi = bias ? bias[n] : 0.0f;
#pragma unroll
    for (int v = 0; v < 8; ++v) {
      int m = m0 + w * 16 + v + 8 * half;
      int idx = m * N + n;
      float val = acc[ct][v] + bi;
      if (accumulate) val += C[idx];
      C[idx] = val;
    }
  }
}

// ---------------------------------------------------------------------------
// Fused pair scorer: rel[b,i,j,:] = relu(relu(left_i+right_j+emb[bucket]) @
//   W_h2 + b_h2) @ W_out + b_out + ss[i] + ss[j]
// One block = (b, 8 i's, 16 j's) -> 128 (i,j) rows; hidden [128x256] stays in
// LDS. Weights arrive pre-swizzled; slab copies are contiguous b128 traffic,
// every WMMA operand is 2x(b128) loads.
// ---------------------------------------------------------------------------
__global__ __launch_bounds__(256) void scorer_kernel(
    const float* __restrict__ leftp, const float* __restrict__ rightp,
    const float* __restrict__ dist_emb, const bf16* __restrict__ wh2sw,
    const float* __restrict__ b_h2, const bf16* __restrict__ wosw,
    const float* __restrict__ b_out, const float* __restrict__ span_scores,
    const int* __restrict__ span_begin, const int* __restrict__ span_end,
    float* __restrict__ rel) {
  extern __shared__ char smem_raw[];
  bf16* lH0 = (bf16*)(smem_raw);             // 128 x 264 rows=(i,j), cols=H
  bf16* lWs = (bf16*)(smem_raw + 67584);     // 16 swizzled W_h2 tiles (16KB)
  bf16* lWo = (bf16*)(smem_raw + 83968);     // all 16 swizzled W_out tiles
  bf16* lH1 = (bf16*)(smem_raw + 100352);    // 128 x 40 relu'd hidden chunk
  int*  lbk = (int*)(smem_raw + 110592);     // 128 buckets

  int t = threadIdx.x;
  int w = t >> 5, lane = t & 31;
  int b  = blockIdx.z;
  int i0 = blockIdx.y * 8;
  int j0 = blockIdx.x * 16;

  // distance buckets per (i,j) pair: d = begin[b,j] - end[b,i]
  if (t < 128) {
    int li = t >> 4, lj = t & 15;
    int dd = span_begin[b * Nn + j0 + lj] - span_end[b * Nn + i0 + li];
    int ad = dd < 0 ? -dd : dd;
    int bk;
    if (ad < 5) {
      bk = ad;
    } else {
      bk = (31 - __clz(ad)) + 3;  // floor(log2(ad)) + 3
      if (bk > 9) bk = 9;
    }
    lbk[t] = bk;
  }
  // stage all of swizzled W_out once (contiguous 16KB copy)
  {
    const v16bf* s = (const v16bf*)wosw;
    v16bf* d = (v16bf*)lWo;
    for (int idx = t; idx < 512; idx += 256) d[idx] = s[idx];
  }
  __syncthreads();

  // h0 = relu(left_i + right_j + emb[bucket]); two rows per pass, each thread
  // produces a packed bf16 channel pair (b64 loads, b32 DS stores).
  for (int pp = 0; pp < 128; pp += 2) {
    int p = pp + (t >> 7);
    int c = (t & 127) * 2;
    int li = p >> 4, lj = p & 15;
    const float* lrow = leftp  + (b * Nn + i0 + li) * Hh + c;
    const float* rrow = rightp + (b * Nn + j0 + lj) * Hh + c;
    const float* erow = dist_emb + lbk[p] * Hh + c;
    v2bf pk;
    pk[0] = (bf16)fmaxf(lrow[0] + rrow[0] + erow[0], 0.0f);
    pk[1] = (bf16)fmaxf(lrow[1] + rrow[1] + erow[1], 0.0f);
    *(v2bf*)(lH0 + p * 264 + c) = pk;
  }
  __syncthreads();

  v8f racc[2];
  racc[0] = zero8();
  racc[1] = zero8();
  int nl = lane & 15, half = lane >> 4;

  for (int nc = 0; nc < 8; ++nc) {  // 32-column chunks of the hidden layer
    // stage W_h2 swizzled tiles for nt = nc*2, nc*2+1 (contiguous 16KB)
    {
      const v16bf* s = (const v16bf*)(wh2sw + nc * 8192);
      v16bf* d = (v16bf*)lWs;
      for (int idx = t; idx < 512; idx += 256) d[idx] = s[idx];
    }
    __syncthreads();

    // GEMM1: h1_chunk = h0 @ W_h2[:, chunk]; wave w owns rows w*16..+15
    v8f acc[2];
    acc[0] = zero8();
    acc[1] = zero8();
    const bf16* lwb = lWs + lane * 16;
#pragma unroll
    for (int kc = 0; kc < 8; ++kc) {
      v16bf av = load_a_rm(lH0 + (w * 16) * 264 + kc * 32, 264, lane);
      acc[0] = wmma_bf16(av, *(const v16bf*)(lwb + (kc << 9)), acc[0]);
      acc[1] = wmma_bf16(av, *(const v16bf*)(lwb + ((8 + kc) << 9)), acc[1]);
    }
    // + b_h2, relu, bf16 into lH1 (wave-local rows; intra-wave RAW only)
#pragma unroll
    for (int ct = 0; ct < 2; ++ct) {
      float bb = b_h2[nc * 32 + ct * 16 + nl];
#pragma unroll
      for (int v = 0; v < 8; ++v) {
        int row = w * 16 + v + 8 * half;
        lH1[row * 40 + ct * 16 + nl] = (bf16)fmaxf(acc[ct][v] + bb, 0.0f);
      }
    }

    // GEMM2: rel += h1_chunk @ W_out[chunk, :]  (K=32 -> one WMMA per tile)
    v16bf a2 = load_a_rm(lH1 + (w * 16) * 40, 40, lane);
    const bf16* lob = lWo + lane * 16;
    racc[0] = wmma_bf16(a2, *(const v16bf*)(lob + (nc << 9)), racc[0]);
    racc[1] = wmma_bf16(a2, *(const v16bf*)(lob + ((8 + nc) << 9)), racc[1]);
    __syncthreads();
  }

  // epilogue: + b_out + pruner scores (both endpoints)
  const float* ss = span_scores + b * Nn;
#pragma unroll
  for (int ct = 0; ct < 2; ++ct) {
    int l = ct * 16 + nl;
    float bo = b_out[l];
#pragma unroll
    for (int v = 0; v < 8; ++v) {
      int p = w * 16 + v + 8 * half;
      int li = p >> 4, lj = p & 15;
      int i = i0 + li, j = j0 + lj;
      rel[((b * Nn + i) * Nn + j) * Ll + l] = racc[ct][v] + bo + ss[i] + ss[j];
    }
  }
}

// ---------------------------------------------------------------------------
// ctxt[b,j,d] = (1/len) * sum_l W_A[l,d] * S[l,d],  S = P^T @ update,
// P[i,l] = relu(rel[b,i,j,l]) * mask[b,i,j].  One block per (b,j).
// A (P^T) transposed through LDS; B tiles stream from the pre-swizzled update
// in global (L2-resident), double-buffered across the ic loop so loads for
// ic+1 overlap the staging barrier + WMMAs of ic.
// ---------------------------------------------------------------------------
__global__ __launch_bounds__(256) void ctxt_kernel(
    const float* __restrict__ rel, const float* __restrict__ mask,
    const bf16* __restrict__ updsw, const float* __restrict__ W_A,
    const int* __restrict__ span_lengths,
    float* __restrict__ ctxt, bf16* __restrict__ ctxb) {
  __shared__ __align__(16) bf16 lA[32 * 40];  // P^T chunk: [l=32][i=32]
  int t = threadIdx.x;
  int w = t >> 5, lane = t & 31;
  int j = blockIdx.x, b = blockIdx.y;

  v8f acc[2][4];
#pragma unroll
  for (int rt = 0; rt < 2; ++rt)
#pragma unroll
    for (int ct = 0; ct < 4; ++ct) acc[rt][ct] = zero8();

  // wave w owns d cols w*64..+63: B tile (ct, ic) at bb + ct*4096 + ic*512
  const bf16* bb = updsw + b * (Nn * Dd) + (((size_t)w * 4 * 8) << 9) + lane * 16;
#define UT(ct, ic) (bb + (ct) * (8 << 9) + (ic) * 512)

  v16bf bc0 = *(const v16bf*)UT(0, 0);
  v16bf bc1 = *(const v16bf*)UT(1, 0);
  v16bf bc2 = *(const v16bf*)UT(2, 0);
  v16bf bc3 = *(const v16bf*)UT(3, 0);

#pragma unroll
  for (int ic = 0; ic < 8; ++ic) {  // K loop over i in chunks of 32
    v16bf bn0 = bc0, bn1 = bc1, bn2 = bc2, bn3 = bc3;
    if (ic + 1 < 8) {  // resolved at compile time (full unroll)
      bn0 = *(const v16bf*)UT(0, ic + 1);
      bn1 = *(const v16bf*)UT(1, ic + 1);
      bn2 = *(const v16bf*)UT(2, ic + 1);
      bn3 = *(const v16bf*)UT(3, ic + 1);
    }
    for (int idx = t; idx < 512; idx += 256) {
      int l = idx & 31, ip = idx >> 5;  // ii pair base
      int i = ic * 32 + ip * 2;
      float m0v = mask[(b * Nn + i) * Nn + j];
      float m1v = mask[(b * Nn + i + 1) * Nn + j];
      float r0 = rel[((b * Nn + i) * Nn + j) * Ll + l];
      float r1 = rel[((b * Nn + i + 1) * Nn + j) * Ll + l];
      v2bf pk;
      pk[0] = (bf16)(fmaxf(r0, 0.0f) * m0v);
      pk[1] = (bf16)(fmaxf(r1, 0.0f) * m1v);
      *(v2bf*)(lA + l * 40 + ip * 2) = pk;
    }
    __syncthreads();
    v16bf a0 = load_a_rm(lA, 40, lane);            // l rows 0..15
    v16bf a1 = load_a_rm(lA + 16 * 40, 40, lane);  // l rows 16..31
    acc[0][0] = wmma_bf16(a0, bc0, acc[0][0]);
    acc[1][0] = wmma_bf16(a1, bc0, acc[1][0]);
    acc[0][1] = wmma_bf16(a0, bc1, acc[0][1]);
    acc[1][1] = wmma_bf16(a1, bc1, acc[1][1]);
    acc[0][2] = wmma_bf16(a0, bc2, acc[0][2]);
    acc[1][2] = wmma_bf16(a1, bc2, acc[1][2]);
    acc[0][3] = wmma_bf16(a0, bc3, acc[0][3]);
    acc[1][3] = wmma_bf16(a1, bc3, acc[1][3]);
    __syncthreads();
    bc0 = bn0; bc1 = bn1; bc2 = bn2; bc3 = bn3;
  }
#undef UT

  float inv = 1.0f / (float)span_lengths[b];
  int nl = lane & 15, half = lane >> 4;
#pragma unroll
  for (int ct = 0; ct < 4; ++ct) {
    int d = w * 64 + ct * 16 + nl;
    float partial = 0.0f;
#pragma unroll
    for (int rt = 0; rt < 2; ++rt)
#pragma unroll
      for (int v = 0; v < 8; ++v)
        partial += acc[rt][ct][v] * W_A[(rt * 16 + v + 8 * half) * Dd + d];
    float tot = partial + __shfl_xor(partial, 16, 32);  // fold partner-lane l's
    if (half == 0) {
      float cv = tot * inv;
      ctxt[(b * Nn + j) * Dd + d] = cv;
      ctxb[(b * Nn + j) * Dd + d] = (bf16)cv;
    }
  }
}

// ---------------------------------------------------------------------------
// Host orchestration
// ---------------------------------------------------------------------------
extern "C" void kernel_launch(void* const* d_in, const int* in_sizes, int n_in,
                              void* d_out, int out_size, void* d_ws, size_t ws_size,
                              hipStream_t stream) {
  (void)in_sizes; (void)n_in; (void)out_size; (void)ws_size;
  const float* all_span_vecs = (const float*)d_in[0];
  const float* span_vecs     = (const float*)d_in[1];
  const float* square_mask   = (const float*)d_in[2];
  const float* span_scores   = (const float*)d_in[3];
  const int*   span_begin    = (const int*)d_in[4];
  const int*   span_end      = (const int*)d_in[5];
  const int*   span_lengths  = (const int*)d_in[6];
  const int*   prune_indices = (const int*)d_in[7];
  const float* W_left  = (const float*)d_in[8];
  const float* b_left  = (const float*)d_in[9];
  const float* W_right = (const float*)d_in[10];
  const float* b_right = (const float*)d_in[11];
  const float* dist_emb= (const float*)d_in[12];
  const float* W_h2    = (const float*)d_in[13];
  const float* b_h2    = (const float*)d_in[14];
  const float* W_out   = (const float*)d_in[15];
  const float* b_out   = (const float*)d_in[16];
  const float* W_A     = (const float*)d_in[17];
  const float* W_gate  = (const float*)d_in[18];
  const float* b_gate  = (const float*)d_in[19];

  float* out_all = (float*)d_out;                 // [B,M,D]
  float* out_upd = out_all + Bb * Mm * Dd;        // [B,N,D]
  float* rel     = out_upd + Bb * Nn * Dd;        // [B,N,N,L] (also working buf)

  // workspace carve-out
  char* ws = (char*)d_ws;
  size_t off = 0;
  auto carve = [&](size_t bytes) {
    char* p = ws + off;
    off += (bytes + 255) & ~(size_t)255;
    return p;
  };
  bf16* wlsw  = (bf16*)carve((size_t)Dd * Hh * sizeof(bf16));   // W_left swizzled
  bf16* wrsw  = (bf16*)carve((size_t)Dd * Hh * sizeof(bf16));   // W_right swizzled
  bf16* wh2sw = (bf16*)carve((size_t)Hh * Hh * sizeof(bf16));   // W_h2 swizzled
  bf16* wosw  = (bf16*)carve((size_t)Hh * Ll * sizeof(bf16));   // W_out swizzled
  bf16* wgtsw = (bf16*)carve((size_t)Dd * Dd * sizeof(bf16));   // W_gate top swz
  bf16* wgbsw = (bf16*)carve((size_t)Dd * Dd * sizeof(bf16));   // W_gate bot swz
  bf16* updb  = (bf16*)carve((size_t)Bb * Nn * Dd * sizeof(bf16));  // row-major
  bf16* ctxb  = (bf16*)carve((size_t)Bb * Nn * Dd * sizeof(bf16));  // row-major
  bf16* updsw = (bf16*)carve((size_t)Bb * Nn * Dd * sizeof(bf16));  // B-swizzled
  float* upd  = (float*)carve((size_t)Bb * Nn * Dd * sizeof(float));
  float* ctxt = (float*)carve((size_t)Bb * Nn * Dd * sizeof(float));
  float* leftb  = (float*)carve((size_t)Bb * Nn * Hh * sizeof(float));
  float* rightb = (float*)carve((size_t)Bb * Nn * Hh * sizeof(float));
  float* gacc   = (float*)carve((size_t)Bb * Nn * Dd * sizeof(float));

  auto swz = [&](const float* s, bf16* dst, int K, int Nc, int batches) {
    dim3 grid((K * Nc + 255) / 256, batches);
    swizzle_b_kernel<<<grid, 256, 0, stream>>>(s, dst, K, Nc);
  };
  swz(W_left,  wlsw,  Dd, Hh, 1);
  swz(W_right, wrsw,  Dd, Hh, 1);
  swz(W_h2,    wh2sw, Hh, Hh, 1);
  swz(W_out,   wosw,  Hh, Ll, 1);
  swz(W_gate,           wgtsw, Dd, Dd, 1);  // top half: multiplies update
  swz(W_gate + Dd * Dd, wgbsw, Dd, Dd, 1);  // bottom half: multiplies ctxt

  const int BND = Bb * Nn * Dd;
  pack_update_kernel<<<BND / 256, 256, 0, stream>>>(span_vecs, upd, updb, BND);
  swz(upd, updsw, Nn, Dd, Bb);  // update as swizzled B-matrix per batch

  auto gemm = [&](const bf16* A, const bf16* Bm, float* C, const float* bias,
                  int M, int N, int accF) {
    dim3 grid(N / 64, M / 128);
    gemm_bf16_k512_kernel<<<grid, 256, 0, stream>>>(A, Bm, C, bias, M, N, accF);
  };
  auto proj = [&]() {  // left/right endpoint projections (K = D = 512)
    gemm(updb, wlsw, leftb,  b_left,  Bb * Nn, Hh, 0);
    gemm(updb, wrsw, rightb, b_right, Bb * Nn, Hh, 0);
  };
  dim3 sgrid(Nn / 16, Nn / 8, Bb);
  const size_t sc_lds = 111104;  // lH0 + lWs + lWo + lH1 + buckets
  auto scorer = [&]() {
    scorer_kernel<<<sgrid, 256, sc_lds, stream>>>(
        leftb, rightb, dist_emb, wh2sw, b_h2, wosw, b_out,
        span_scores, span_begin, span_end, rel);
  };

  proj();
  scorer();  // rel0

  for (int it = 0; it < 2; ++it) {  // REL_PROP
    ctxt_kernel<<<dim3(Nn, Bb), 256, 0, stream>>>(
        rel, square_mask, updsw, W_A, span_lengths, ctxt, ctxb);
    gemm(updb, wgtsw, gacc, nullptr, Bb * Nn, Dd, 0);  // update @ Wg_top
    gemm(ctxb, wgbsw, gacc, nullptr, Bb * Nn, Dd, 1);  // += ctxt @ Wg_bot
    gate_finish_kernel<<<BND / 256, 256, 0, stream>>>(
        gacc, b_gate, upd, ctxt, upd, updb, BND);
    swz(upd, updsw, Nn, Dd, Bb);  // refresh swizzled update for next ctxt
    proj();
    scorer();  // rel_{it+1}
  }

  // outputs: all_out (copy + scatter), update, rel (already in place)
  copy_f32_kernel<<<(Bb * Mm * Dd) / 256, 256, 0, stream>>>(all_span_vecs, out_all, Bb * Mm * Dd);
  scatter_kernel<<<Bb * Nn, 256, 0, stream>>>(out_all, upd, prune_indices, span_lengths);
  copy_f32_kernel<<<BND / 256, 256, 0, stream>>>(upd, out_upd, BND);
}
